// MoELayer_71648644432052
// MI455X (gfx1250) — matrix operand order repaired and checked by reading, not compiled, not run on hardware
//
#include <hip/hip_runtime.h>
#include <hip/hip_bf16.h>

// ---------------- static problem config ----------------
#define T_TOK   4096          // 2 * 2048 tokens
#define DM      1024          // d_model
#define DFF     2048          // d_ff
#define NE      8             // experts
#define CAP_ROWS 8320         // 8192 pair-rows + 8 experts * 15 alignment pad
#define MAX_TILES 520         // CAP_ROWS / 16
#define XS_STR  (DM + 8)      // LDS row stride (halves) - avoid bank conflicts
#define HS_STR  (DFF + 8)
#define SMEM_BYTES ((16 * XS_STR + 16 * HS_STR) * 2)

// swizzled-weight region (bf16, WMMA B-fragment order)
#define WSW_BASE   139264ull                       // 256-aligned, past small region
#define WSW_ONE    (33554432ull)                   // 8*1024*2048 halves * 2B
#define WSW_NEED   (WSW_BASE + 3ull * WSW_ONE)     // ~96.2 MB

typedef __attribute__((ext_vector_type(16))) __bf16 v16bf;
typedef __attribute__((ext_vector_type(8)))  float  v8f;

__device__ __forceinline__ float fast_silu_mul(float g, float u) {
    // silu(g) * u with single v_rcp_f32 instead of IEEE division
    return g * __builtin_amdgcn_rcpf(1.0f + __expf(-g)) * u;
}

// ---------------- init: zero output + counters, default bucket ----------------
__global__ void moe_k_init(float* __restrict__ out, int* __restrict__ cnt,
                           float* __restrict__ imp, float* __restrict__ loadcnt,
                           float* __restrict__ zacc, int* __restrict__ bucketTok,
                           float* __restrict__ bucketWgt) {
    size_t gid = (size_t)blockIdx.x * 256 + threadIdx.x;
    if (gid < (size_t)T_TOK * DM) out[gid] = 0.0f;
    if (gid < CAP_ROWS) { bucketTok[gid] = 0; bucketWgt[gid] = 0.0f; }
    if (gid < NE) { cnt[gid] = 0; imp[gid] = 0.0f; loadcnt[gid] = 0.0f; }
    if (gid == 0) zacc[0] = 0.0f;
}

// ---------------- router: one wave32 per token ----------------
__global__ __launch_bounds__(256) void moe_k_router(
    const float* __restrict__ x, const float* __restrict__ rw,
    int* __restrict__ routeIdx, float* __restrict__ routeW,
    int* __restrict__ cnt, float* __restrict__ imp,
    float* __restrict__ loadcnt, float* __restrict__ zacc) {
    const int wave = threadIdx.x >> 5;
    const int lane = threadIdx.x & 31;
    const int t = blockIdx.x * 8 + wave;
    if (t >= T_TOK) return;

    float acc[NE];
#pragma unroll
    for (int e = 0; e < NE; ++e) acc[e] = 0.0f;
    const float* xr = x + (size_t)t * DM;
    for (int d = lane; d < DM; d += 32) {
        float xv = xr[d];
#pragma unroll
        for (int e = 0; e < NE; ++e) acc[e] += xv * rw[d * NE + e];
    }
#pragma unroll
    for (int e = 0; e < NE; ++e)
        for (int off = 16; off > 0; off >>= 1)
            acc[e] += __shfl_xor(acc[e], off, 32);

    if (lane == 0) {
        float mx = acc[0];
#pragma unroll
        for (int e = 1; e < NE; ++e) mx = fmaxf(mx, acc[e]);
        float p[NE], den = 0.0f, zs = 0.0f;
#pragma unroll
        for (int e = 0; e < NE; ++e) {
            zs += acc[e] * acc[e];
            p[e] = __expf(acc[e] - mx);
            den += p[e];
        }
        float inv = 1.0f / den;
#pragma unroll
        for (int e = 0; e < NE; ++e) p[e] *= inv;
        // top-2 (ties -> lowest index, matching lax.top_k)
        int i0 = 0;
#pragma unroll
        for (int e = 1; e < NE; ++e) if (p[e] > p[i0]) i0 = e;
        int i1 = (i0 == 0) ? 1 : 0;
#pragma unroll
        for (int e = 0; e < NE; ++e) if (e != i0 && p[e] > p[i1]) i1 = e;
        float s = p[i0] + p[i1];
        routeIdx[2 * t]     = i0;  routeIdx[2 * t + 1] = i1;
        routeW[2 * t]       = p[i0] / s;
        routeW[2 * t + 1]   = p[i1] / s;
        atomicAdd(&cnt[i0], 1);
        atomicAdd(&cnt[i1], 1);
#pragma unroll
        for (int e = 0; e < NE; ++e) atomicAdd(&imp[e], p[e]);
        atomicAdd(&loadcnt[i0], 1.0f);
        atomicAdd(zacc, zs);
    }
}

// ---------------- offsets + tile map + loss finalize (single thread) ----------------
__global__ void moe_k_offsets(const int* __restrict__ cnt, int* __restrict__ alignedOff,
                              int* __restrict__ fill, int* __restrict__ tileExpert,
                              const float* __restrict__ imp, const float* __restrict__ loadcnt,
                              const float* __restrict__ zacc, float* __restrict__ outTail) {
    int off = 0;
    for (int e = 0; e < NE; ++e) {
        alignedOff[e] = off;
        fill[e] = off;
        int tiles = (cnt[e] + 15) >> 4;
        for (int tt = 0; tt < tiles; ++tt) tileExpert[(off >> 4) + tt] = e;
        off += tiles << 4;
    }
    alignedOff[NE] = off;
    for (int tt = off >> 4; tt < MAX_TILES; ++tt) tileExpert[tt] = 0;
    float aux = 0.0f;
    const float invT = 1.0f / (float)T_TOK;
    for (int e = 0; e < NE; ++e) aux += (imp[e] * invT) * (loadcnt[e] * invT);
    outTail[0] = (float)NE * aux * 0.01f;                       // aux_loss
    outTail[1] = (zacc[0] / (float)(T_TOK * NE)) * 0.001f;      // z_loss
}

// ---------------- scatter tokens into per-expert buckets ----------------
__global__ void moe_k_scatter(const int* __restrict__ routeIdx, const float* __restrict__ routeW,
                              int* __restrict__ fill, int* __restrict__ bucketTok,
                              float* __restrict__ bucketWgt) {
    int t = blockIdx.x * blockDim.x + threadIdx.x;
    if (t >= T_TOK) return;
#pragma unroll
    for (int s = 0; s < 2; ++s) {
        int e = routeIdx[2 * t + s];
        int pos = atomicAdd(&fill[e], 1);
        bucketTok[pos] = t;
        bucketWgt[pos] = routeW[2 * t + s];
    }
}

// ---------------- weight pre-swizzle: fp32 [K][N] -> bf16 B-fragment order ----------------
// dst layout (halves): (((e * N/16 + ntile) * K/32) + kblock) * 512 + lane*16 + j
// where lane: n = ntile*16 + (lane&15), krow = kblock*32 + (lane>>4)*16, j = 0..15
__global__ __launch_bounds__(256) void moe_k_swizzle(
    const float* __restrict__ src, __bf16* __restrict__ dst, int K, int N) {
    const int fid  = blockIdx.x * 8 + (threadIdx.x >> 5);
    const int lane = threadIdx.x & 31;
    const int kb   = K >> 5;                // k-blocks per column tile
    const int fpe  = (N >> 4) * kb;         // frags per expert
    const int e    = fid / fpe;
    const int f    = fid % fpe;
    const int ntile = f / kb;
    const int kb0   = f % kb;
    const int n    = ntile * 16 + (lane & 15);
    const int krow = kb0 * 32 + (lane >> 4) * 16;
    const float* s = src + (size_t)e * K * N;
    alignas(16) __bf16 h[16];
#pragma unroll
    for (int j = 0; j < 16; ++j)
        h[j] = (__bf16)s[(size_t)(krow + j) * N + n];
    __bf16* d = dst + (size_t)fid * 512 + lane * 16;
    *(uint4*)(d)     = *(const uint4*)&h[0];
    *(uint4*)(d + 8) = *(const uint4*)&h[8];
}

// ---------------- fused expert MLP (FAST: pre-swizzled bf16 weights) ----------------
__global__ __launch_bounds__(256) void moe_k_expert_pre(
    const float* __restrict__ x,
    const __bf16* __restrict__ w1b, const __bf16* __restrict__ w2b,
    const __bf16* __restrict__ w3b,
    const int* __restrict__ bucketTok, const float* __restrict__ bucketWgt,
    const int* __restrict__ tileExpert, float* __restrict__ out) {
    extern __shared__ char smem[];
    __bf16* Xs = (__bf16*)smem;                        // [16][XS_STR]
    __bf16* Hs = (__bf16*)(smem + 16 * XS_STR * 2);    // [16][HS_STR]
    __shared__ int   tt[16];
    __shared__ float tw[16];

    const int e    = tileExpert[blockIdx.x];
    const int base = blockIdx.x * 16;
    if (threadIdx.x < 16) {
        tt[threadIdx.x] = bucketTok[base + threadIdx.x];
        tw[threadIdx.x] = bucketWgt[base + threadIdx.x];
    }
    __syncthreads();

    // Phase 0: gather + pre-scale tokens into LDS as bf16
    for (int idx = threadIdx.x; idx < 16 * DM; idx += 256) {
        int r = idx >> 10, c = idx & (DM - 1);
        float w = tw[r];
        float v = (w != 0.0f) ? x[(size_t)tt[r] * DM + c] * w : 0.0f;
        Xs[r * XS_STR + c] = (__bf16)v;
    }
    __syncthreads();

    const int wave = threadIdx.x >> 5;
    const int lane = threadIdx.x & 31;
    const int hn   = lane & 15;
    const int hk   = lane >> 4;
    const __bf16* xrowBase = Xs + hn * XS_STR;

    // Phase 1: gate/up GEMMs + SwiGLU -> Hs[16][2048]
    {
        const size_t fpeFF = (size_t)(DFF / 16) * (DM / 32);   // 4096 frags/expert
        const __bf16* W1B = w1b + (size_t)e * fpeFF * 512;
        const __bf16* W3B = w3b + (size_t)e * fpeFF * 512;
        for (int nt = 0; nt < 16; ++nt) {
            const int ntileIdx = wave * 16 + nt;
            const int ncol = ntileIdx * 16 + hn;
            const __bf16* bp1 = W1B + (size_t)ntileIdx * (DM / 32) * 512 + lane * 16;
            const __bf16* bp3 = W3B + (size_t)ntileIdx * (DM / 32) * 512 + lane * 16;
            v8f accg = {}; v8f accu = {};
#pragma unroll 2
            for (int kb0 = 0; kb0 < DM / 32; ++kb0) {
                v16bf a;
                const __bf16* xr = xrowBase + kb0 * 32 + hk * 16;
#pragma unroll
                for (int j = 0; j < 8; ++j) { a[j] = xr[j]; a[8 + j] = xr[8 + j]; }
                v16bf b1 = *(const v16bf*)(bp1 + (size_t)kb0 * 512);
                v16bf b3 = *(const v16bf*)(bp3 + (size_t)kb0 * 512);
                accg = __builtin_amdgcn_wmma_f32_16x16x32_bf16(false, a, false, b1,
                                                               (short)0, accg, false, false);
                accu = __builtin_amdgcn_wmma_f32_16x16x32_bf16(false, a, false, b3,
                                                               (short)0, accu, false, false);
            }
#pragma unroll
            for (int i = 0; i < 8; ++i) {
                int m = hk * 8 + i;
                Hs[m * HS_STR + ncol] = (__bf16)fast_silu_mul(accg[i], accu[i]);
            }
        }
    }
    __syncthreads();

    // Phase 2: down-projection [16,2048] x [2048,1024] -> atomic scatter-add
    {
        const size_t fpeDM = (size_t)(DM / 16) * (DFF / 32);   // 4096 frags/expert
        const __bf16* W2B = w2b + (size_t)e * fpeDM * 512;
        const __bf16* hrowBase = Hs + hn * HS_STR;
        for (int jt = 0; jt < 8; ++jt) {
            const int ntileIdx = wave * 8 + jt;
            const int ncol = ntileIdx * 16 + hn;
            const __bf16* bp = W2B + (size_t)ntileIdx * (DFF / 32) * 512 + lane * 16;
            v8f acc = {};
#pragma unroll 2
            for (int kb0 = 0; kb0 < DFF / 32; ++kb0) {
                v16bf a;
                const __bf16* hr = hrowBase + kb0 * 32 + hk * 16;
#pragma unroll
                for (int j = 0; j < 8; ++j) { a[j] = hr[j]; a[8 + j] = hr[8 + j]; }
                v16bf b = *(const v16bf*)(bp + (size_t)kb0 * 512);
                acc = __builtin_amdgcn_wmma_f32_16x16x32_bf16(false, a, false, b,
                                                              (short)0, acc, false, false);
            }
#pragma unroll
            for (int i = 0; i < 8; ++i) {
                int m = hk * 8 + i;
                if (tw[m] != 0.0f)
                    atomicAdd(&out[(size_t)tt[m] * DM + ncol], acc[i]);
            }
        }
    }
}

// ---------------- fused expert MLP (FALLBACK: inline fp32->bf16 convert) ----------------
__global__ __launch_bounds__(256) void moe_k_expert(
    const float* __restrict__ x, const float* __restrict__ w1,
    const float* __restrict__ w2, const float* __restrict__ w3,
    const int* __restrict__ bucketTok, const float* __restrict__ bucketWgt,
    const int* __restrict__ tileExpert, float* __restrict__ out) {
    extern __shared__ char smem[];
    __bf16* Xs = (__bf16*)smem;
    __bf16* Hs = (__bf16*)(smem + 16 * XS_STR * 2);
    __shared__ int   tt[16];
    __shared__ float tw[16];

    const int e    = tileExpert[blockIdx.x];
    const int base = blockIdx.x * 16;
    if (threadIdx.x < 16) {
        tt[threadIdx.x] = bucketTok[base + threadIdx.x];
        tw[threadIdx.x] = bucketWgt[base + threadIdx.x];
    }
    __syncthreads();

    for (int idx = threadIdx.x; idx < 16 * DM; idx += 256) {
        int r = idx >> 10, c = idx & (DM - 1);
        float w = tw[r];
        float v = (w != 0.0f) ? x[(size_t)tt[r] * DM + c] * w : 0.0f;
        Xs[r * XS_STR + c] = (__bf16)v;
    }
    __syncthreads();

    const int wave = threadIdx.x >> 5;
    const int lane = threadIdx.x & 31;
    const int hn   = lane & 15;
    const int hk   = lane >> 4;
    const __bf16* xrowBase = Xs + hn * XS_STR;

    {
        const size_t eb = (size_t)e * DM * DFF;
        const float* W1 = w1 + eb;
        const float* W3 = w3 + eb;
        for (int nt = 0; nt < 16; ++nt) {
            const int ncol = wave * 256 + nt * 16 + hn;
            v8f accg = {}; v8f accu = {};
            for (int k0 = 0; k0 < DM; k0 += 32) {
                v16bf a;
                const __bf16* xr = xrowBase + k0 + hk * 16;
#pragma unroll
                for (int j = 0; j < 8; ++j) { a[j] = xr[j]; a[8 + j] = xr[8 + j]; }
                const int krow = k0 + hk * 16;
                v16bf b1, b3;
#pragma unroll
                for (int j = 0; j < 16; ++j) {
                    size_t off = (size_t)(krow + j) * DFF + ncol;
                    b1[j] = (__bf16)W1[off];
                    b3[j] = (__bf16)W3[off];
                }
                accg = __builtin_amdgcn_wmma_f32_16x16x32_bf16(false, a, false, b1,
                                                               (short)0, accg, false, false);
                accu = __builtin_amdgcn_wmma_f32_16x16x32_bf16(false, a, false, b3,
                                                               (short)0, accu, false, false);
            }
#pragma unroll
            for (int i = 0; i < 8; ++i) {
                int m = hk * 8 + i;
                Hs[m * HS_STR + ncol] = (__bf16)fast_silu_mul(accg[i], accu[i]);
            }
        }
    }
    __syncthreads();

    {
        const float* W2 = w2 + (size_t)e * DFF * DM;
        const __bf16* hrowBase = Hs + hn * HS_STR;
        for (int jt = 0; jt < 8; ++jt) {
            const int ncol = (wave * 8 + jt) * 16 + hn;
            v8f acc = {};
            for (int k0 = 0; k0 < DFF; k0 += 32) {
                v16bf a;
                const __bf16* hr = hrowBase + k0 + hk * 16;
#pragma unroll
                for (int j = 0; j < 8; ++j) { a[j] = hr[j]; a[8 + j] = hr[8 + j]; }
                const int krow = k0 + hk * 16;
                v16bf b;
#pragma unroll
                for (int j = 0; j < 16; ++j)
                    b[j] = (__bf16)W2[(size_t)(krow + j) * DM + ncol];
                acc = __builtin_amdgcn_wmma_f32_16x16x32_bf16(false, a, false, b,
                                                              (short)0, acc, false, false);
            }
#pragma unroll
            for (int i = 0; i < 8; ++i) {
                int m = hk * 8 + i;
                if (tw[m] != 0.0f)
                    atomicAdd(&out[(size_t)tt[m] * DM + ncol], acc[i]);
            }
        }
    }
}

extern "C" void kernel_launch(void* const* d_in, const int* in_sizes, int n_in,
                              void* d_out, int out_size, void* d_ws, size_t ws_size,
                              hipStream_t stream) {
    const float* x  = (const float*)d_in[0];
    const float* rw = (const float*)d_in[1];
    const float* w1 = (const float*)d_in[2];
    const float* w2 = (const float*)d_in[3];
    const float* w3 = (const float*)d_in[4];
    float* out = (float*)d_out;

    char* ws = (char*)d_ws;
    int*   cnt        = (int*)  (ws + 0);
    int*   alignedOff = (int*)  (ws + 256);
    float* imp        = (float*)(ws + 512);
    float* loadcnt    = (float*)(ws + 768);
    float* zacc       = (float*)(ws + 1024);
    int*   fill       = (int*)  (ws + 1280);
    int*   tileExpert = (int*)  (ws + 1536);
    int*   routeIdx   = (int*)  (ws + 4096);
    float* routeW     = (float*)(ws + 36864);
    int*   bucketTok  = (int*)  (ws + 69632);
    float* bucketWgt  = (float*)(ws + 102912);
    __bf16* w1b = (__bf16*)(ws + WSW_BASE);
    __bf16* w3b = (__bf16*)(ws + WSW_BASE + WSW_ONE);
    __bf16* w2b = (__bf16*)(ws + WSW_BASE + 2 * WSW_ONE);

    const bool fast = (ws_size >= WSW_NEED);

    moe_k_init<<<(T_TOK * DM + 255) / 256, 256, 0, stream>>>(
        out, cnt, imp, loadcnt, zacc, bucketTok, bucketWgt);

    if (fast) {
        // 32768 frags per tensor / 8 frags per block = 4096 blocks
        moe_k_swizzle<<<4096, 256, 0, stream>>>(w1, w1b, DM, DFF);
        moe_k_swizzle<<<4096, 256, 0, stream>>>(w3, w3b, DM, DFF);
        moe_k_swizzle<<<4096, 256, 0, stream>>>(w2, w2b, DFF, DM);
    }

    moe_k_router<<<T_TOK / 8, 256, 0, stream>>>(
        x, rw, routeIdx, routeW, cnt, imp, loadcnt, zacc);

    moe_k_offsets<<<1, 1, 0, stream>>>(
        cnt, alignedOff, fill, tileExpert, imp, loadcnt, zacc,
        out + (size_t)T_TOK * DM);

    moe_k_scatter<<<(T_TOK + 255) / 256, 256, 0, stream>>>(
        routeIdx, routeW, fill, bucketTok, bucketWgt);

    if (fast) {
        moe_k_expert_pre<<<MAX_TILES, 256, SMEM_BYTES, stream>>>(
            x, w1b, w2b, w3b, bucketTok, bucketWgt, tileExpert, out);
    } else {
        moe_k_expert<<<MAX_TILES, 256, SMEM_BYTES, stream>>>(
            x, w1, w2, w3, bucketTok, bucketWgt, tileExpert, out);
    }
}
